// Block_22385369546994
// MI455X (gfx1250) — matrix operand order repaired and checked by reading, not compiled, not run on hardware
//
#include <hip/hip_runtime.h>
#include <hip/hip_bf16.h>

// ---------------------------------------------------------------------------
// Types for CDNA5 WMMA (gfx1250, wave32)
// ---------------------------------------------------------------------------
typedef __attribute__((ext_vector_type(16))) __bf16 v16bf;
typedef __attribute__((ext_vector_type(8)))  float  v8f;

union FragBF { v16bf v; unsigned u[8]; };

__device__ __forceinline__ v8f wmma_bf16(const FragBF& a, const FragBF& b, v8f c) {
    // D(f32 16x16) = A(bf16 16x32) * B(bf16 32x16) + C
    return __builtin_amdgcn_wmma_f32_16x16x32_bf16(
        /*neg_a=*/false, a.v, /*neg_b=*/false, b.v,
        /*c_mod=*/(short)0, c, /*reuse_a=*/false, /*reuse_b=*/false);
}

__device__ __forceinline__ unsigned short f2bf(float x) {
    unsigned u = __builtin_bit_cast(unsigned, x);
    unsigned r = u + 0x7FFFu + ((u >> 16) & 1u);   // round-to-nearest-even
    return (unsigned short)(r >> 16);
}

// ---------------------------------------------------------------------------
// Problem constants
// ---------------------------------------------------------------------------
#define BB  4
#define NN  1024
#define CC  1024
#define HH  16
#define HD  64
#define HID 4096
#define ROWS (BB * NN)          // 4096

// ---------------------------------------------------------------------------
// fp32 [K,N] -> bf16 [N,K] (transpose so WMMA B-fragments are dword loads)
// ---------------------------------------------------------------------------
__global__ void __launch_bounds__(256) k_transpose_bf16(
    const float* __restrict__ in, unsigned short* __restrict__ out, int K, int N) {
    long long idx = (long long)blockIdx.x * 256 + threadIdx.x;
    long long total = (long long)K * N;
    if (idx >= total) return;
    int k = (int)(idx / N), n = (int)(idx % N);
    out[(long long)n * K + k] = f2bf(in[idx]);
}

// ---------------------------------------------------------------------------
// LayerNorm over C=1024, fp32 in -> bf16 out.  One 256-thread block per row.
// ---------------------------------------------------------------------------
__global__ void __launch_bounds__(256) k_layernorm(
    const float* __restrict__ x, const float* __restrict__ g,
    const float* __restrict__ b, unsigned short* __restrict__ out) {
    int row = blockIdx.x;
    const float* xr = x + (long long)row * CC;
    float vals[4], s = 0.f, s2 = 0.f;
#pragma unroll
    for (int i = 0; i < 4; ++i) {
        int c = threadIdx.x + i * 256;
        float v = xr[c];
        vals[i] = v; s += v; s2 += v * v;
    }
#pragma unroll
    for (int off = 16; off; off >>= 1) {
        s  += __shfl_xor(s,  off, 32);
        s2 += __shfl_xor(s2, off, 32);
    }
    __shared__ float sh1[8], sh2[8];
    int w = threadIdx.x >> 5;
    if ((threadIdx.x & 31) == 0) { sh1[w] = s; sh2[w] = s2; }
    __syncthreads();
    s = 0.f; s2 = 0.f;
#pragma unroll
    for (int i = 0; i < 8; ++i) { s += sh1[i]; s2 += sh2[i]; }
    float mean = s * (1.0f / CC);
    float var  = s2 * (1.0f / CC) - mean * mean;
    float inv  = rsqrtf(var + 1e-5f);
    unsigned short* orow = out + (long long)row * CC;
#pragma unroll
    for (int i = 0; i < 4; ++i) {
        int c = threadIdx.x + i * 256;
        orow[c] = f2bf((vals[i] - mean) * inv * g[c] + b[c]);
    }
}

// ---------------------------------------------------------------------------
// Generic bf16 WMMA GEMM: out[M,N] = A[M,K] * Bt[N,K]^T  (+bias,+gelu,+res)
// One wave computes a 64x64 tile (4x4 WMMA accumulators), K stepped by 32:
// 8 fragment loads feed 16 WMMAs per step (~33 flop/byte operand reuse).
// A-frag (16x32): lane m=l&15; VGPR j: k=kh*16+2j(+1).
// B-frag (32x16): lane n=l&15; VGPR j: k=kh*16+2j(+1)  (kh = l>>4).
// ---------------------------------------------------------------------------
template<bool DO_BIAS, bool DO_RES, bool DO_GELU, bool OUT_BF16>
__global__ void __launch_bounds__(256) k_gemm_bf16(
    const unsigned short* __restrict__ A,   // [M,K] bf16
    const unsigned short* __restrict__ Bt,  // [N,K] bf16 (transposed weights)
    const float* __restrict__ bias, const float* __restrict__ res,
    void* __restrict__ out, int M, int N, int K) {
    int wid  = blockIdx.x * 8 + (threadIdx.x >> 5);
    int lane = threadIdx.x & 31;
    int Nt = N >> 6, Mt = M >> 6;
    if (wid >= Mt * Nt) return;              // grids sized exactly; whole-wave
    int mt = wid / Nt, nt = wid % Nt;
    int MB = mt * 64, NB = nt * 64;
    int ln = lane & 15, kh = lane >> 4;

    v8f acc[4][4] = {};
    const unsigned short* ap[4];
    const unsigned short* bp[4];
#pragma unroll
    for (int i = 0; i < 4; ++i) {
        ap[i] = A  + (long long)(MB + 16 * i + ln) * K;
        bp[i] = Bt + (long long)(NB + 16 * i + ln) * K;
    }

    for (int kk = 0; kk < K; kk += 32) {
        FragBF af[4], bfm[4];
#pragma unroll
        for (int mi = 0; mi < 4; ++mi) {
#pragma unroll
            for (int j = 0; j < 8; ++j) {
                int k1 = kk + kh * 16 + 2 * j;
                af[mi].u[j]     = *(const unsigned*)(ap[mi] + k1);
            }
        }
#pragma unroll
        for (int ni = 0; ni < 4; ++ni) {
            const unsigned short* bq = bp[ni] + kk + kh * 16;
#pragma unroll
            for (int j = 0; j < 8; ++j)
                bfm[ni].u[j] = *(const unsigned*)(bq + 2 * j);
        }
        if (kk + 32 < K) {                   // uniform: prefetch next K-tile
#pragma unroll
            for (int i = 0; i < 4; ++i) {
                __builtin_prefetch(ap[i] + kk + 32 + kh * 16, 0, 3);
                __builtin_prefetch(bp[i] + kk + 32 + kh * 16, 0, 3);
            }
        }
#pragma unroll
        for (int mi = 0; mi < 4; ++mi)
#pragma unroll
            for (int ni = 0; ni < 4; ++ni)
                acc[mi][ni] = wmma_bf16(af[mi], bfm[ni], acc[mi][ni]);
    }

    // Epilogue.  C-layout: VGPR j -> row m = j + 8*kh ; lane -> col n = ln.
#pragma unroll
    for (int mi = 0; mi < 4; ++mi)
#pragma unroll
        for (int ni = 0; ni < 4; ++ni)
#pragma unroll
            for (int j = 0; j < 8; ++j) {
                int m = MB + mi * 16 + j + 8 * kh;
                int n = NB + ni * 16 + ln;
                float v = acc[mi][ni][j];
                if (DO_BIAS) v += bias[n];
                if (DO_GELU) v = 0.5f * v * (1.0f + erff(v * 0.70710678118f));
                if (DO_RES)  v += res[(long long)m * N + n];
                if (OUT_BF16)
                    ((unsigned short*)out)[(long long)m * N + n] = f2bf(v);
                else
                    ((float*)out)[(long long)m * N + n] = v;
            }
}

// ---------------------------------------------------------------------------
// RoPE + head split.  qkv fp32 [ROWS, 3C] ->
//   q bf16 [B,H,N,HD] (pre-scaled by HD^-0.5), k bf16 [B,H,N,HD],
//   v bf16 transposed [B,H,HD,N]  (so attention B-fragments are dword loads)
// One thread per (b,h,n,pair).
// ---------------------------------------------------------------------------
__global__ void __launch_bounds__(256) k_rope_split(
    const float* __restrict__ qkv, const float* __restrict__ fcos,
    const float* __restrict__ fsin, unsigned short* __restrict__ qo,
    unsigned short* __restrict__ ko, unsigned short* __restrict__ vt) {
    int t = blockIdx.x * 256 + threadIdx.x;       // B*H*N*32 = 2097152
    int d = t & 31;
    int n = (t >> 5) & (NN - 1);
    int h = (t >> 15) & (HH - 1);
    int b = t >> 19;
    long long row = (long long)b * NN + n;
    const float* qk = qkv + row * (3 * CC);
    float c = fcos[row * 32 + d], s = fsin[row * 32 + d];
    int col = h * HD + 2 * d;
    float qr = qk[col],          qi = qk[col + 1];
    float kr = qk[CC + col],     ki = qk[CC + col + 1];
    long long ob = (((long long)(b * HH + h)) * NN + n) * HD + 2 * d;
    const float scale = 0.125f;                   // HD^-0.5
    qo[ob]     = f2bf((qr * c - qi * s) * scale);
    qo[ob + 1] = f2bf((qr * s + qi * c) * scale);
    ko[ob]     = f2bf(kr * c - ki * s);
    ko[ob + 1] = f2bf(kr * s + ki * c);
    float v0 = qk[2 * CC + col], v1 = qk[2 * CC + col + 1];
    long long vb = ((long long)(b * HH + h) * HD + 2 * d) * NN + n;
    vt[vb]      = f2bf(v0);
    vt[vb + NN] = f2bf(v1);
}

// ---------------------------------------------------------------------------
// Flash attention: one wave per (b, h, 16-query tile). 32-key chunks:
//   S = Q*K^T via 4 WMMAs, online softmax (shfl row reductions),
//   P staged through LDS (bf16) to remap C-layout -> A-fragment,
//   O += P*V via 4 WMMAs.  Output bf16 [B,N,H*HD].
// ---------------------------------------------------------------------------
__global__ void __launch_bounds__(256) k_attention(
    const unsigned short* __restrict__ q, const unsigned short* __restrict__ k,
    const unsigned short* __restrict__ vt, unsigned short* __restrict__ o) {
    __shared__ unsigned short ldsP[8][16 * 32];
    int lane = threadIdx.x & 31;
    int wv   = threadIdx.x >> 5;
    int wid  = blockIdx.x * 8 + wv;               // 4096 waves total
    int qt = wid & 63;
    int h  = (wid >> 6) & (HH - 1);
    int b  = wid >> 10;
    int ln = lane & 15, kh = lane >> 4;

    const unsigned short* qh = q  + ((long long)(b * HH + h)) * NN * HD;
    const unsigned short* kp = k  + ((long long)(b * HH + h)) * NN * HD;
    const unsigned short* vh = vt + ((long long)(b * HH + h)) * HD * NN;
    unsigned short* ph = ldsP[wv];

    // Q A-fragments: two K=32 steps across HD=64.
    FragBF qa[2];
    {
        const unsigned short* qrow = qh + (long long)(qt * 16 + ln) * HD;
#pragma unroll
        for (int t = 0; t < 2; ++t)
#pragma unroll
            for (int j = 0; j < 8; ++j) {
                int k1 = t * 32 + kh * 16 + 2 * j;
                qa[t].u[j]     = *(const unsigned*)(qrow + k1);
            }
    }

    v8f oacc[4] = {};
    float rmax[8], rsum[8];
#pragma unroll
    for (int j = 0; j < 8; ++j) { rmax[j] = -1e30f; rsum[j] = 0.f; }

    for (int kc = 0; kc < NN / 32; ++kc) {
        // ---- S = Q * K^T for 2 sub-tiles of 16 keys ----
        v8f S[2];
#pragma unroll
        for (int t2 = 0; t2 < 2; ++t2) {
            int key = kc * 32 + t2 * 16 + ln;     // B-frag col n = key
            const unsigned short* krow = kp + (long long)key * HD;
            v8f s = {};
#pragma unroll
            for (int t = 0; t < 2; ++t) {
                FragBF bfm;
#pragma unroll
                for (int j = 0; j < 8; ++j)       // k(hd) = t*32 + kh*16 + 2j
                    bfm.u[j] = *(const unsigned*)(krow + t * 32 + kh * 16 + 2 * j);
                s = wmma_bf16(qa[t], bfm, s);
            }
            S[t2] = s;
        }
        // ---- online softmax: row m = j + 8*kh lives across a 16-lane group ----
        float scl[8];
#pragma unroll
        for (int j = 0; j < 8; ++j) {
            float mx = fmaxf(S[0][j], S[1][j]);
#pragma unroll
            for (int off = 8; off; off >>= 1) mx = fmaxf(mx, __shfl_xor(mx, off, 16));
            float nm = fmaxf(rmax[j], mx);
            float sc = __expf(rmax[j] - nm);
            float p0 = __expf(S[0][j] - nm);
            float p1 = __expf(S[1][j] - nm);
            float rs = p0 + p1;
#pragma unroll
            for (int off = 8; off; off >>= 1) rs += __shfl_xor(rs, off, 16);
            rsum[j] = rsum[j] * sc + rs;
            rmax[j] = nm;
            scl[j] = sc;
            int m = j + 8 * kh;
            ph[m * 32 + ln]      = f2bf(p0);      // P tile 16x32 bf16 in LDS
            ph[m * 32 + 16 + ln] = f2bf(p1);
        }
#pragma unroll
        for (int ni = 0; ni < 4; ++ni)
#pragma unroll
            for (int j = 0; j < 8; ++j)
                oacc[ni][j] *= scl[j];
        asm volatile("s_wait_dscnt 0" ::: "memory");   // own-wave LDS RAW
        // ---- P as A-fragment (16x32) ----
        FragBF pa;
#pragma unroll
        for (int j = 0; j < 8; ++j) {
            int k1 = kh * 16 + 2 * j;
            pa.u[j]     = *(const unsigned*)(ph + ln * 32 + k1);
        }
        // ---- O += P * V  (V transposed: [HD, N]) ----
#pragma unroll
        for (int ni = 0; ni < 4; ++ni) {
            const unsigned short* vrow =
                vh + (long long)(ni * 16 + ln) * NN + kc * 32 + kh * 16;
            FragBF vb;
#pragma unroll
            for (int j = 0; j < 8; ++j)
                vb.u[j] = *(const unsigned*)(vrow + 2 * j);
            oacc[ni] = wmma_bf16(pa, vb, oacc[ni]);
        }
    }
    // ---- finalize: /rowsum, write bf16 [B,N,H*HD] ----
    unsigned short* orow = o + ((long long)b * NN + qt * 16) * CC + h * HD;
#pragma unroll
    for (int ni = 0; ni < 4; ++ni)
#pragma unroll
        for (int j = 0; j < 8; ++j) {
            int m = j + 8 * kh;
            orow[(long long)m * CC + ni * 16 + ln] = f2bf(oacc[ni][j] / rsum[j]);
        }
}

// ---------------------------------------------------------------------------
// Host-side orchestration
// ---------------------------------------------------------------------------
extern "C" void kernel_launch(void* const* d_in, const int* in_sizes, int n_in,
                              void* d_out, int out_size, void* d_ws, size_t ws_size,
                              hipStream_t stream) {
    const float* x     = (const float*)d_in[0];
    const float* fcos  = (const float*)d_in[1];
    const float* fsin  = (const float*)d_in[2];
    const float* ln1_g = (const float*)d_in[3];
    const float* ln1_b = (const float*)d_in[4];
    const float* w_qkv = (const float*)d_in[5];
    const float* w_prj = (const float*)d_in[6];
    const float* b_prj = (const float*)d_in[7];
    const float* ln2_g = (const float*)d_in[8];
    const float* ln2_b = (const float*)d_in[9];
    const float* w_fc1 = (const float*)d_in[10];
    const float* b_fc1 = (const float*)d_in[11];
    const float* w_fc2 = (const float*)d_in[12];
    const float* b_fc2 = (const float*)d_in[13];
    float* out = (float*)d_out;

    // ---- workspace layout ----
    char* w = (char*)d_ws;
    float* qkv           = (float*)w;          w += (long long)ROWS * 3 * CC * 4;   // 48 MB
    float* x1            = (float*)w;          w += (long long)ROWS * CC * 4;       // 16 MB
    unsigned short* hbuf = (unsigned short*)w; w += (long long)ROWS * CC * 2;       //  8 MB
    unsigned short* wqkvT= (unsigned short*)w; w += (long long)3 * CC * CC * 2;     //  6 MB
    unsigned short* wprjT= (unsigned short*)w; w += (long long)CC * CC * 2;         //  2 MB
    unsigned short* wfc1T= (unsigned short*)w; w += (long long)HID * CC * 2;        //  8 MB
    unsigned short* wfc2T= (unsigned short*)w; w += (long long)CC * HID * 2;        //  8 MB
    unsigned short* qb   = (unsigned short*)w; w += (long long)ROWS * CC * 2;       //  8 MB
    unsigned short* kb   = (unsigned short*)w; w += (long long)ROWS * CC * 2;       //  8 MB
    unsigned short* vtb  = (unsigned short*)w; w += (long long)ROWS * CC * 2;       //  8 MB
    unsigned short* ob   = (unsigned short*)w; w += (long long)ROWS * CC * 2;       //  8 MB
    unsigned short* h2   = (unsigned short*)w; w += (long long)ROWS * HID * 2;      // 32 MB

    // ---- weight transpose+convert to bf16 ----
    k_transpose_bf16<<<(CC * 3 * CC + 255) / 256, 256, 0, stream>>>(w_qkv, wqkvT, CC, 3 * CC);
    k_transpose_bf16<<<(CC * CC + 255) / 256,     256, 0, stream>>>(w_prj, wprjT, CC, CC);
    k_transpose_bf16<<<(CC * HID + 255) / 256,    256, 0, stream>>>(w_fc1, wfc1T, CC, HID);
    k_transpose_bf16<<<(HID * CC + 255) / 256,    256, 0, stream>>>(w_fc2, wfc2T, HID, CC);

    // ---- LN1 -> h ----
    k_layernorm<<<ROWS, 256, 0, stream>>>(x, ln1_g, ln1_b, hbuf);

    // ---- QKV GEMM: [4096,1024] x [1024,3072] -> fp32 ----
    k_gemm_bf16<false, false, false, false>
        <<<(ROWS / 64) * (3 * CC / 64) / 8, 256, 0, stream>>>(
            hbuf, wqkvT, nullptr, nullptr, qkv, ROWS, 3 * CC, CC);

    // ---- RoPE + head split ----
    k_rope_split<<<(BB * HH * NN * 32) / 256, 256, 0, stream>>>(qkv, fcos, fsin, qb, kb, vtb);

    // ---- flash attention (4096 waves) ----
    k_attention<<<BB * HH * (NN / 16) / 8, 256, 0, stream>>>(qb, kb, vtb, ob);

    // ---- proj GEMM + bias + residual(x) -> x1 fp32 ----
    k_gemm_bf16<true, true, false, false>
        <<<(ROWS / 64) * (CC / 64) / 8, 256, 0, stream>>>(
            ob, wprjT, b_prj, x, x1, ROWS, CC, CC);

    // ---- LN2 -> h ----
    k_layernorm<<<ROWS, 256, 0, stream>>>(x1, ln2_g, ln2_b, hbuf);

    // ---- FC1 GEMM + bias + exact GELU -> bf16 h2 ----
    k_gemm_bf16<true, false, true, true>
        <<<(ROWS / 64) * (HID / 64) / 8, 256, 0, stream>>>(
            hbuf, wfc1T, b_fc1, nullptr, h2, ROWS, HID, CC);

    // ---- FC2 GEMM + bias + residual(x1) -> d_out fp32 ----
    k_gemm_bf16<true, true, false, false>
        <<<(ROWS / 64) * (CC / 64) / 8, 256, 0, stream>>>(
            h2, wfc2T, b_fc2, x1, out, ROWS, CC, HID);
}
